// SimpleDownBlock_49478023250389
// MI455X (gfx1250) — matrix operand order repaired and checked by reading, not compiled, not run on hardware
//
#include <hip/hip_runtime.h>

#define RG 64
#define NI 64
#define NF 128
#define NEMBD 256
#define NBATCH 4

typedef __attribute__((ext_vector_type(16))) _Float16 v16h;
typedef __attribute__((ext_vector_type(8)))  float    v8f;

// ---------------------------------------------------------------- async LDS staging
// CDNA5 async global->LDS copy (ASYNCcnt-tracked), one 16B chunk per lane.
// VDST = wave-relative LDS byte address = low 32 bits of the flat shared pointer
// (flat->LDS aperture mapping truncates to addr[31:0]).
__device__ __forceinline__ void async_ld_lds_b128(void* lds_dst, const void* gsrc) {
    unsigned loff = (unsigned)(size_t)lds_dst;
    asm volatile("global_load_async_to_lds_b128 %0, %1, off"
                 :: "v"(loff), "v"(gsrc)
                 : "memory");
}

__device__ __forceinline__ void wait_async_zero() {
#if __has_builtin(__builtin_amdgcn_s_wait_asynccnt)
    __builtin_amdgcn_s_wait_asynccnt(0);
#else
    asm volatile("s_wait_asynccnt 0x0" ::: "memory");
#endif
}

// ---------------------------------------------------------------- utilities
__global__ void fill_neg1_kernel(int* __restrict__ g, int n) {
    int i = blockIdx.x * blockDim.x + threadIdx.x;
    if (i < n) g[i] = -1;
}

__global__ void zero_f_kernel(float* __restrict__ p, int n) {
    int i = blockIdx.x * blockDim.x + threadIdx.x;
    if (i < n) p[i] = 0.0f;
}

__global__ void scatter_kernel(const int* __restrict__ coords, int* __restrict__ g, int N) {
    int i = blockIdx.x * blockDim.x + threadIdx.x;
    if (i >= N) return;
    int b = coords[4 * i + 0], x = coords[4 * i + 1];
    int y = coords[4 * i + 2], z = coords[4 * i + 3];
    g[((b * RG + x) * RG + y) * RG + z] = i;
}

// emb[b][c] = sum_e silu(t[b,e]) * Wt[e,c] + bt[c]   (one block of 256 threads)
__global__ void emb_kernel(const float* __restrict__ t, const float* __restrict__ Wt,
                           const float* __restrict__ bt, float* __restrict__ emb) {
    int i = threadIdx.x;               // 0..255
    int b = i >> 6, c = i & 63;
    float s = bt[c];
    for (int e = 0; e < NEMBD; ++e) {
        float x  = t[b * NEMBD + e];
        float sl = x / (1.0f + expf(-x));
        s += sl * Wt[e * NI + c];
    }
    emb[i] = s;
}

// Pack W[k][c][d] (c = K in-channel (64), d = N out-channel (Cout)) into a contiguous-k (k = 16h + j)
// V_WMMA_F32_16X16X32_F16 B-operand lane layout:
//   dst[(((k*tiles+t)*2+q)*32 + L)*16 + j] = W[k][32q + (L<16 ? j : 16+j)][16t + (L&15)]
__global__ void pack_w_kernel(const float* __restrict__ W, _Float16* __restrict__ dst,
                              int tiles, int Cout, int total) {
    int i = blockIdx.x * blockDim.x + threadIdx.x;   // one per (k,t,q,L)
    if (i >= total) return;
    int L  = i & 31;
    int q  = (i >> 5) & 1;
    int kt = i >> 6;
    int t  = kt % tiles;
    int k  = kt / tiles;
    int d    = 16 * t + (L & 15);
    int cbas = 32 * q + ((L < 16) ? 0 : 16);
    _Float16* o = dst + (size_t)i * 16;
#pragma unroll
    for (int j = 0; j < 16; ++j)
        o[j] = (_Float16)W[((size_t)(k * NI + cbas + j)) * Cout + d];
}

// ---------------------------------------------------------------- fused gather+GEMM conv
// 128 threads = 4 waves; 64 points per block (16 per wave). Per tap:
//   * all 128 threads async-copy the tap's packed B block (TILES*2KB) into LDS
//   * 16 lanes/wave do the voxel-hash neighbor lookup for their wave's rows
//   * all lanes gather the neighbor feature rows (+optional time-emb) into the
//     wave's f16 A tile, row-major, read by lanes in contiguous k = 16h + s order
//   * 2 K-chunks x TILES WMMAs per wave, B read from the shared LDS stage
template <int TAPS, int TILES, bool EMBADD, bool STRIDE2>
__global__ __launch_bounds__(128) void conv_kernel(
    const float* __restrict__ src, const int* __restrict__ crds,
    const int* __restrict__ grid, const float* __restrict__ emb,
    const _Float16* __restrict__ Wpk, float* __restrict__ out, int NROWS) {

    constexpr int COUT   = TILES * 16;
    constexpr int WBLOCK = TILES * 2 * 32 * 16;   // halves per tap

    __shared__ __align__(16) _Float16 Wt_s[WBLOCK];
    __shared__ __align__(16) _Float16 At[4][16][80];
    __shared__ int nidx[4][16];
    __shared__ int bat[4][16];

    const int tid  = threadIdx.x;
    const int wave = tid >> 5;
    const int lane = tid & 31;
    const int p0w  = blockIdx.x * 64 + wave * 16;

    int cb_ = 0, cx = 0, cy = 0, cz = 0;
    bool rowok = false;
    if (lane < 16) {
        int p = p0w + lane;
        rowok = (p < NROWS);
        if (rowok) {
            cb_ = crds[4 * p + 0];
            cx  = crds[4 * p + 1];
            cy  = crds[4 * p + 2];
            cz  = crds[4 * p + 3];
            if (STRIDE2) { cx *= 2; cy *= 2; cz *= 2; }
        }
        bat[wave][lane] = cb_;
    }

    v8f acc[TILES];
#pragma unroll
    for (int t = 0; t < TILES; ++t)
#pragma unroll
        for (int v = 0; v < 8; ++v) acc[t][v] = 0.0f;

    for (int k = 0; k < TAPS; ++k) {
        // ---- async-stage this tap's packed B block into LDS (all 128 threads)
        const _Float16* wsrc = Wpk + (size_t)k * WBLOCK;
#pragma unroll
        for (int u = 0; u < TILES; ++u) {
            int idx = (u * 128 + tid) * 8;            // 8 halves = 16B per chunk
            async_ld_lds_b128(&Wt_s[idx], wsrc + idx);
        }
        // ---- neighbor lookup for this wave's 16 rows
        int dx, dy, dz;
        if (STRIDE2) { dx = (k >> 2) & 1; dy = (k >> 1) & 1; dz = k & 1; }
        else         { dx = k / 9 - 1;    dy = (k / 3) % 3 - 1; dz = k % 3 - 1; }
        if (lane < 16) {
            int id = -1;
            if (rowok) {
                int nx = cx + dx, ny = cy + dy, nz = cz + dz;
                if (((unsigned)nx < (unsigned)RG) & ((unsigned)ny < (unsigned)RG) &
                    ((unsigned)nz < (unsigned)RG))
                    id = grid[((cb_ * RG + nx) * RG + ny) * RG + nz];
            }
            nidx[wave][lane] = id;
        }
        __syncthreads();
        // ---- gather row r (lane>>1), 32-col half (lane&1) into this wave's A tile
        {
            const int r = lane >> 1, c0 = (lane & 1) * 32;
            const int id = nidx[wave][r];
            const float* erow = EMBADD ? (emb + bat[wave][r] * NI + c0) : (const float*)0;
            const float* srow = src + (size_t)(id < 0 ? 0 : id) * NI + c0;
#pragma unroll
            for (int u = 0; u < 32; u += 4) {
                float4 v = make_float4(0.f, 0.f, 0.f, 0.f);
                if (id >= 0) {
                    v = *(const float4*)(srow + u);
                    if (EMBADD) {
                        float4 e = *(const float4*)(erow + u);
                        v = make_float4(v.x + e.x, v.y + e.y, v.z + e.z, v.w + e.w);
                    }
                }
                At[wave][r][c0 + u + 0] = (_Float16)v.x;
                At[wave][r][c0 + u + 1] = (_Float16)v.y;
                At[wave][r][c0 + u + 2] = (_Float16)v.z;
                At[wave][r][c0 + u + 3] = (_Float16)v.w;
            }
        }
        wait_async_zero();          // own wave's staging chunks landed in LDS
        __syncthreads();            // everyone's chunks + A tiles visible
        // ---- WMMAs: A from this wave's tile, B from the shared LDS stage
        const int m  = lane & 15;
        const int hb = (lane < 16) ? 0 : 16;
#pragma unroll
        for (int q = 0; q < 2; ++q) {
            v16h a;
            const int cb = 32 * q + hb;
#pragma unroll
            for (int j = 0; j < 8; ++j) a[j] = At[wave][m][cb + j];
#pragma unroll
            for (int j = 0; j < 8; ++j) a[8 + j] = At[wave][m][cb + 8 + j];
#pragma unroll
            for (int t = 0; t < TILES; ++t) {
                const _Float16* bp = &Wt_s[((t * 2 + q) * 32 + lane) * 16];
                v16h bv;
#pragma unroll
                for (int j = 0; j < 16; ++j) bv[j] = bp[j];
                acc[t] = __builtin_amdgcn_wmma_f32_16x16x32_f16(
                    false, a, false, bv, (short)0, acc[t], false, false);
            }
        }
        __syncthreads();            // protect Wt_s / A tiles before next tap
    }
    // D layout: elem v, lane L -> row M = v + 8*(L>=16), col = 16t + (L&15)
    const int n0 = lane & 15;
    const int mo = (lane < 16) ? 0 : 8;
#pragma unroll
    for (int t = 0; t < TILES; ++t)
#pragma unroll
        for (int v = 0; v < 8; ++v) {
            int row = p0w + v + mo;
            if (row < NROWS) out[(size_t)row * COUT + t * 16 + n0] = acc[t][v];
        }
}

// ---------------------------------------------------------------- BN helpers
// sums layout at base: [sum(C) | sumsq(C)]  (pre-zeroed)
__global__ void reduce_stats_kernel(const float* __restrict__ x, float* __restrict__ sums,
                                    int rows, int C) {
    __shared__ float sb[256];
    __shared__ float sb2[256];
    const int tid = threadIdx.x;
    const int c   = tid % C;
    const int rpb = 256 / C;
    float s = 0.f, s2 = 0.f;
    for (int r = blockIdx.x * rpb + tid / C; r < rows; r += gridDim.x * rpb) {
        float v = x[(size_t)r * C + c];
        s += v; s2 += v * v;
    }
    sb[tid] = s; sb2[tid] = s2;
    __syncthreads();
    if (tid < C) {
        float a = 0.f, b = 0.f;
        for (int u = tid; u < 256; u += C) { a += sb[u]; b += sb2[u]; }
        atomicAdd(&sums[c], a);
        atomicAdd(&sums[C + c], b);
    }
}

__global__ void bn_relu_kernel(const float* __restrict__ xin, float* __restrict__ xout,
                               const float* __restrict__ sums, const float* __restrict__ g,
                               const float* __restrict__ bta, int rows, int C) {
    long long i = (long long)blockIdx.x * blockDim.x + threadIdx.x;
    if (i >= (long long)rows * C) return;
    int c = (int)(i % C);
    float inv = 1.0f / (float)rows;
    float m   = sums[c] * inv;
    float var = sums[C + c] * inv - m * m;
    float y   = (xin[i] - m) * rsqrtf(var + 1e-5f) * g[c] + bta[c];
    xout[i]   = y > 0.f ? y : 0.f;
}

// ---------------------------------------------------------------- launch
extern "C" void kernel_launch(void* const* d_in, const int* in_sizes, int n_in,
                              void* d_out, int out_size, void* d_ws, size_t ws_size,
                              hipStream_t stream) {
    const float* feats = (const float*)d_in[0];
    const float* t     = (const float*)d_in[1];
    const int*   crd   = (const int*)  d_in[2];
    const int*   cds   = (const int*)  d_in[3];
    const float* Wt    = (const float*)d_in[4];
    const float* bt    = (const float*)d_in[5];
    const float* W1    = (const float*)d_in[6];
    const float* g1    = (const float*)d_in[7];
    const float* b1    = (const float*)d_in[8];
    const float* W2    = (const float*)d_in[9];
    const float* g2    = (const float*)d_in[10];
    const float* b2    = (const float*)d_in[11];
    (void)n_in; (void)out_size; (void)ws_size;

    const int N = in_sizes[0] / NI;   // points
    const int M = in_sizes[3] / 4;    // downsampled coords

    char* ws = (char*)d_ws;
    size_t off = 0;
    auto alloc = [&](size_t bytes) -> void* {
        void* p = ws + off;
        off += (bytes + 255) & ~(size_t)255;
        return p;
    };
    int*      grid  = (int*)     alloc((size_t)NBATCH * RG * RG * RG * sizeof(int));
    float*    emb   = (float*)   alloc((size_t)NBATCH * NI * sizeof(float));
    float*    sums  = (float*)   alloc((size_t)(2 * NI + 2 * NF) * sizeof(float));
    _Float16* Wp1   = (_Float16*)alloc((size_t)27 * NI * NI * sizeof(_Float16));
    _Float16* Wp2   = (_Float16*)alloc((size_t)8 * NI * NF * sizeof(_Float16));
    float*    hbuf  = (float*)   alloc((size_t)N * NI * sizeof(float));
    float*    oraw  = (float*)   alloc((size_t)M * NF * sizeof(float));
    float*    sums1 = sums;
    float*    sums2 = sums + 2 * NI;

    const int gridN = NBATCH * RG * RG * RG;
    fill_neg1_kernel<<<(gridN + 255) / 256, 256, 0, stream>>>(grid, gridN);
    scatter_kernel  <<<(N + 255) / 256, 256, 0, stream>>>(crd, grid, N);
    zero_f_kernel   <<<2, 256, 0, stream>>>(sums, 2 * NI + 2 * NF);
    emb_kernel      <<<1, 256, 0, stream>>>(t, Wt, bt, emb);

    const int tot1 = 27 * 4 * 2 * 32;
    const int tot2 = 8 * 8 * 2 * 32;
    pack_w_kernel<<<(tot1 + 255) / 256, 256, 0, stream>>>(W1, Wp1, 4, NI, tot1);
    pack_w_kernel<<<(tot2 + 255) / 256, 256, 0, stream>>>(W2, Wp2, 8, NF, tot2);

    conv_kernel<27, 4, true, false><<<(N + 63) / 64, 128, 0, stream>>>(
        feats, crd, grid, emb, Wp1, hbuf, N);
    reduce_stats_kernel<<<256, 256, 0, stream>>>(hbuf, sums1, N, NI);
    bn_relu_kernel<<<(int)(((long long)N * NI + 255) / 256), 256, 0, stream>>>(
        hbuf, hbuf, sums1, g1, b1, N, NI);

    conv_kernel<8, 8, false, true><<<(M + 63) / 64, 128, 0, stream>>>(
        hbuf, cds, grid, emb, Wp2, oraw, M);
    reduce_stats_kernel<<<256, 256, 0, stream>>>(oraw, sums2, M, NF);
    bn_relu_kernel<<<(int)(((long long)M * NF + 255) / 256), 256, 0, stream>>>(
        oraw, (float*)d_out, sums2, g2, b2, M, NF);
}
